// KV_Extension_ClearCLIP_54279796687482
// MI455X (gfx1250) — matrix operand order repaired and checked by reading, not compiled, not run on hardware
//
#include <hip/hip_runtime.h>
#include <hip/hip_bf16.h>

typedef __attribute__((ext_vector_type(16))) _Float16 v16h;
typedef __attribute__((ext_vector_type(8)))  _Float16 v8h;
typedef __attribute__((ext_vector_type(8)))  float    v8f;

#define N_TOK  4096
#define C_FEAT 768
#define S_TOK  1024
#define NH     12
#define DH     64
#define NCLUS  32
#define TIN    24
#define HW     32

// ---------------------------------------------------------------------------
// Kernel 1: bilinear resize v_ext [48,24,24,64] -> vT (f16) [12][64][4096]
// half-pixel centers, edge clamp (matches jax.image.resize bilinear upscale)
// ---------------------------------------------------------------------------
__global__ void k_resize(const float* __restrict__ vext, _Float16* __restrict__ vT) {
    int x  = blockIdx.x;      // 0..31
    int y  = blockIdx.y;      // 0..31
    int bh = blockIdx.z;      // 0..47
    int d  = threadIdx.x;     // 0..63
    int b = bh / NH, h = bh % NH;

    float fx = (x + 0.5f) * 0.75f - 0.5f;
    float fy = (y + 0.5f) * 0.75f - 0.5f;
    int x0 = (int)floorf(fx), y0 = (int)floorf(fy);
    float wx = fx - (float)x0, wy = fy - (float)y0;
    int x0c = max(x0, 0), x1c = min(x0 + 1, TIN - 1);
    int y0c = max(y0, 0), y1c = min(y0 + 1, TIN - 1);

    const float* base = vext + (size_t)bh * TIN * TIN * DH;
    float v00 = base[(y0c * TIN + x0c) * DH + d];
    float v01 = base[(y0c * TIN + x1c) * DH + d];
    float v10 = base[(y1c * TIN + x0c) * DH + d];
    float v11 = base[(y1c * TIN + x1c) * DH + d];
    float val = (1.f - wy) * ((1.f - wx) * v00 + wx * v01)
              +        wy  * ((1.f - wx) * v10 + wx * v11);

    int n = b * S_TOK + y * HW + x;
    vT[(size_t)(h * DH + d) * N_TOK + n] = (_Float16)val;
}

// ---------------------------------------------------------------------------
// Kernel 2: per-cluster sums of x.  One block per cluster, no atomics.
// csum[c][0..767] = sum over rows i with idx[i]==c of x[i][:]
// ---------------------------------------------------------------------------
__global__ void k_cluster_sum(const float* __restrict__ x, const int* __restrict__ idx,
                              float* __restrict__ csum) {
    __shared__ int sidx[N_TOK];
    int c = blockIdx.x;
    int t = threadIdx.x;     // 0..255
    for (int i = t; i < N_TOK; i += 256) sidx[i] = idx[i];
    __syncthreads();

    float a0 = 0.f, a1 = 0.f, a2 = 0.f;
    for (int i = 0; i < N_TOK; ++i) {
        if (sidx[i] == c) {
            const float* r = x + (size_t)i * C_FEAT;
            a0 += r[t]; a1 += r[t + 256]; a2 += r[t + 512];
        }
    }
    float* o = csum + (size_t)c * C_FEAT;
    o[t] = a0; o[t + 256] = a1; o[t + 512] = a2;
}

// ---------------------------------------------------------------------------
// Kernel 3: proxy[i] = csum[ind[i]] + sum_{8-neighbors j, ind[j]!=ind[i]} x[j]
// then l2-normalize proxy and x[i]; write f16 proxy_n / key_n. Block per row.
// (count divisor cancels under l2norm since it is a positive scalar)
// ---------------------------------------------------------------------------
__global__ void k_proxy_key(const float* __restrict__ x, const int* __restrict__ idx,
                            const float* __restrict__ csum,
                            _Float16* __restrict__ proxy_n, _Float16* __restrict__ key_n) {
    __shared__ float red[256];
    int i = blockIdx.x, t = threadIdx.x;
    int ci = idx[i];

    float p0 = csum[(size_t)ci * C_FEAT + t];
    float p1 = csum[(size_t)ci * C_FEAT + t + 256];
    float p2 = csum[(size_t)ci * C_FEAT + t + 512];

    int blk = i / S_TOK, s = i % S_TOK;
    int r = s / HW, cx = s % HW;
    #pragma unroll
    for (int dr = -1; dr <= 1; ++dr) {
        #pragma unroll
        for (int dc = -1; dc <= 1; ++dc) {
            if (dr == 0 && dc == 0) continue;
            int nr = r + dr, nc = cx + dc;
            if (nr < 0 || nr >= HW || nc < 0 || nc >= HW) continue;
            int j = blk * S_TOK + nr * HW + nc;
            if (idx[j] != ci) {
                const float* xr = x + (size_t)j * C_FEAT;
                p0 += xr[t]; p1 += xr[t + 256]; p2 += xr[t + 512];
            }
        }
    }

    red[t] = p0 * p0 + p1 * p1 + p2 * p2;
    __syncthreads();
    for (int o = 128; o > 0; o >>= 1) { if (t < o) red[t] += red[t + o]; __syncthreads(); }
    float inv = 1.0f / fmaxf(sqrtf(red[0]), 1e-12f);
    __syncthreads();
    proxy_n[(size_t)i * C_FEAT + t]       = (_Float16)(p0 * inv);
    proxy_n[(size_t)i * C_FEAT + t + 256] = (_Float16)(p1 * inv);
    proxy_n[(size_t)i * C_FEAT + t + 512] = (_Float16)(p2 * inv);

    const float* xr = x + (size_t)i * C_FEAT;
    float k0 = xr[t], k1 = xr[t + 256], k2 = xr[t + 512];
    red[t] = k0 * k0 + k1 * k1 + k2 * k2;
    __syncthreads();
    for (int o = 128; o > 0; o >>= 1) { if (t < o) red[t] += red[t + o]; __syncthreads(); }
    inv = 1.0f / fmaxf(sqrtf(red[0]), 1e-12f);
    key_n[(size_t)i * C_FEAT + t]       = (_Float16)(k0 * inv);
    key_n[(size_t)i * C_FEAT + t + 256] = (_Float16)(k1 * inv);
    key_n[(size_t)i * C_FEAT + t + 512] = (_Float16)(k2 * inv);
}

// ---------------------------------------------------------------------------
// WMMA fragment loaders (ISA 7.12.2, wave32).
// A 16x32 f16: lane (r=lane&15, hi=lane>>4) holds row r, K = hi*16..+15
//              contiguous -> two 16B loads.
// B 32x16 f16 from N-major buffer Bt[N][K]: lane holds col r, K = hi*16..+15
//              contiguous -> one 32B load.
// ---------------------------------------------------------------------------
__device__ __forceinline__ v16h ld_frag_a(const _Float16* __restrict__ base,
                                          int row0, int k0, int ld, int lane) {
    int r = lane & 15, hi = lane >> 4;
    const _Float16* p = base + (size_t)(row0 + r) * ld + k0 + hi * 16;
    v8h lo = *(const v8h*)p;
    v8h hh = *(const v8h*)(p + 8);
    v16h o;
    #pragma unroll
    for (int j = 0; j < 8; ++j) { o[j] = lo[j]; o[j + 8] = hh[j]; }
    return o;
}

__device__ __forceinline__ v16h ld_frag_b(const _Float16* __restrict__ baseT,
                                          int col0, int k0, int ld, int lane) {
    int r = lane & 15, hi = lane >> 4;
    const _Float16* p = baseT + (size_t)(col0 + r) * ld + k0 + hi * 16;
    return *(const v16h*)p;
}

// ---------------------------------------------------------------------------
// Kernel 4: attn_raw = proxy_n @ key_n^T   (f16 in, f32 out)  M=N=4096 K=768
// wave tile 32x64 (2x4 accumulators), block = 4 waves -> 32x256 per block
// ---------------------------------------------------------------------------
__global__ void k_gemm_qk(const _Float16* __restrict__ A, const _Float16* __restrict__ Bt,
                          float* __restrict__ D) {
    int lane = threadIdx.x & 31;
    int wave = threadIdx.x >> 5;
    int row0 = blockIdx.y * 32;
    int col0 = blockIdx.x * 256 + wave * 64;

    v8f acc[2][4];
    #pragma unroll
    for (int mt = 0; mt < 2; ++mt)
        #pragma unroll
        for (int nt = 0; nt < 4; ++nt)
            acc[mt][nt] = (v8f){0.f,0.f,0.f,0.f,0.f,0.f,0.f,0.f};

    for (int k0 = 0; k0 < C_FEAT; k0 += 32) {
        if (k0 + 32 < C_FEAT) {
            __builtin_prefetch(A  + (size_t)(row0 + (lane & 15)) * C_FEAT + k0 + 32, 0, 1);
            __builtin_prefetch(Bt + (size_t)(col0 + (lane & 15)) * C_FEAT + k0 + 32, 0, 1);
        }
        v16h a0 = ld_frag_a(A, row0,      k0, C_FEAT, lane);
        v16h a1 = ld_frag_a(A, row0 + 16, k0, C_FEAT, lane);
        v16h bf[4];
        #pragma unroll
        for (int nt = 0; nt < 4; ++nt) bf[nt] = ld_frag_b(Bt, col0 + 16 * nt, k0, C_FEAT, lane);
        #pragma unroll
        for (int nt = 0; nt < 4; ++nt) {
            acc[0][nt] = __builtin_amdgcn_wmma_f32_16x16x32_f16(false, a0, false, bf[nt],
                                                                (short)0, acc[0][nt], false, false);
            acc[1][nt] = __builtin_amdgcn_wmma_f32_16x16x32_f16(false, a1, false, bf[nt],
                                                                (short)0, acc[1][nt], false, false);
        }
    }

    int hi = lane >> 4, lo = lane & 15;
    #pragma unroll
    for (int mt = 0; mt < 2; ++mt)
        #pragma unroll
        for (int nt = 0; nt < 4; ++nt)
            #pragma unroll
            for (int v = 0; v < 8; ++v) {
                int rr = row0 + 16 * mt + hi * 8 + v;
                int cc = col0 + 16 * nt + lo;
                D[(size_t)rr * N_TOK + cc] = acc[mt][nt][v];
            }
}

// ---------------------------------------------------------------------------
// Kernel 5: per-row token_norm + dynamic cut + masked softmax -> P (f16)
// one 256-thread block per row; 16 elements/thread in registers
// ---------------------------------------------------------------------------
__global__ void k_softmax(const float* __restrict__ attn, _Float16* __restrict__ P) {
    __shared__ float rsum[256];
    __shared__ float rmax[256];
    int r = blockIdx.x, t = threadIdx.x;
    const float* row = attn + (size_t)r * N_TOK;

    float a[16];
    float s = 0.f, m = -1e30f;
    #pragma unroll
    for (int j = 0; j < 16; ++j) {
        a[j] = row[t + 256 * j];
        s += a[j];
        m = fmaxf(m, a[j]);
    }
    rsum[t] = s; rmax[t] = m;
    __syncthreads();
    for (int o = 128; o > 0; o >>= 1) {
        if (t < o) { rsum[t] += rsum[t + o]; rmax[t] = fmaxf(rmax[t], rmax[t + o]); }
        __syncthreads();
    }
    float mean  = rsum[0] * (1.0f / (float)N_TOK);
    float shift = 1.2f * fmaxf(mean, 0.f);            // beta * relu(mean)
    float mx    = (rmax[0] - shift) * 3.0f;           // gamma
    float cut   = fminf(mx, 0.1f);                    // cutting_hp
    __syncthreads();

    float e[16];
    float es = 0.f;
    #pragma unroll
    for (int j = 0; j < 16; ++j) {
        float ap = (a[j] - shift) * 3.0f;
        float ev = (ap < cut) ? 0.f : __expf((ap - mx) * (1.0f / 0.07f));
        e[j] = ev; es += ev;
    }
    rsum[t] = es;
    __syncthreads();
    for (int o = 128; o > 0; o >>= 1) { if (t < o) rsum[t] += rsum[t + o]; __syncthreads(); }
    float inv = 1.0f / rsum[0];

    _Float16* prow = P + (size_t)r * N_TOK;
    #pragma unroll
    for (int j = 0; j < 16; ++j) prow[t + 256 * j] = (_Float16)(e[j] * inv);
}

// ---------------------------------------------------------------------------
// Kernel 6: out = P @ v per (b,h).  M=1024 N=64 K=4096 per GEMM, 48 GEMMs.
// wave tile 32x64; direct store into out[s][b][h*64+d] (f32)
// ---------------------------------------------------------------------------
__global__ void k_gemm_pv(const _Float16* __restrict__ P, const _Float16* __restrict__ vT,
                          float* __restrict__ out) {
    int lane = threadIdx.x & 31;
    int wave = threadIdx.x >> 5;
    int bh = blockIdx.x;                    // 0..47
    int b = bh / NH, h = bh % NH;
    int m0 = (blockIdx.y * 4 + wave) * 32;  // row within this b's 1024

    const _Float16* A  = P  + (size_t)(b * S_TOK) * N_TOK;
    const _Float16* Bt = vT + (size_t)h * DH * N_TOK;

    v8f acc[2][4];
    #pragma unroll
    for (int mt = 0; mt < 2; ++mt)
        #pragma unroll
        for (int nt = 0; nt < 4; ++nt)
            acc[mt][nt] = (v8f){0.f,0.f,0.f,0.f,0.f,0.f,0.f,0.f};

    for (int k0 = 0; k0 < N_TOK; k0 += 32) {
        if (k0 + 32 < N_TOK) {
            __builtin_prefetch(A  + (size_t)(m0 + (lane & 15)) * N_TOK + k0 + 32, 0, 1);
            __builtin_prefetch(Bt + (size_t)(lane & 15) * N_TOK + k0 + 32, 0, 1);
        }
        v16h a0 = ld_frag_a(A, m0,      k0, N_TOK, lane);
        v16h a1 = ld_frag_a(A, m0 + 16, k0, N_TOK, lane);
        v16h bf[4];
        #pragma unroll
        for (int nt = 0; nt < 4; ++nt) bf[nt] = ld_frag_b(Bt, 16 * nt, k0, N_TOK, lane);
        #pragma unroll
        for (int nt = 0; nt < 4; ++nt) {
            acc[0][nt] = __builtin_amdgcn_wmma_f32_16x16x32_f16(false, a0, false, bf[nt],
                                                                (short)0, acc[0][nt], false, false);
            acc[1][nt] = __builtin_amdgcn_wmma_f32_16x16x32_f16(false, a1, false, bf[nt],
                                                                (short)0, acc[1][nt], false, false);
        }
    }

    int hi = lane >> 4, lo = lane & 15;
    #pragma unroll
    for (int mt = 0; mt < 2; ++mt)
        #pragma unroll
        for (int nt = 0; nt < 4; ++nt)
            #pragma unroll
            for (int v = 0; v < 8; ++v) {
                int s = m0 + 16 * mt + hi * 8 + v;     // 0..1023
                int d = 16 * nt + lo;                  // 0..63
                out[((size_t)s * 4 + b) * (NH * DH) + h * DH + d] = acc[mt][nt][v];
            }
}

// ---------------------------------------------------------------------------
// Workspace layout (bytes):
//   [0,   64MB)  attn   f32 [4096][4096]
//   [64,  96MB)  P      f16 [4096][4096]
//   [96, 102MB)  proxy_n f16 [4096][768]
//   [102,108MB)  key_n   f16 [4096][768]
//   [108,114MB)  vT      f16 [12][64][4096]
//   [114MB, +96KB) csum  f32 [32][768]
// ---------------------------------------------------------------------------
extern "C" void kernel_launch(void* const* d_in, const int* in_sizes, int n_in,
                              void* d_out, int out_size, void* d_ws, size_t ws_size,
                              hipStream_t stream) {
    const float* x    = (const float*)d_in[0];   // ex_feats_grid -> [4096][768]
    const float* vext = (const float*)d_in[1];   // [48][24][24][64]
    const int*   idx  = (const int*)d_in[2];     // [4096]
    float* out = (float*)d_out;                  // [1024][4][768]

    char* ws = (char*)d_ws;
    float*    attn    = (float*)   (ws);
    _Float16* P       = (_Float16*)(ws + ((size_t)64  << 20));
    _Float16* proxy_n = (_Float16*)(ws + ((size_t)96  << 20));
    _Float16* key_n   = (_Float16*)(ws + ((size_t)102 << 20));
    _Float16* vT      = (_Float16*)(ws + ((size_t)108 << 20));
    float*    csum    = (float*)   (ws + ((size_t)114 << 20));

    k_resize     <<<dim3(HW, HW, 4 * NH), DH,  0, stream>>>(vext, vT);
    k_cluster_sum<<<NCLUS,                256, 0, stream>>>(x, idx, csum);
    k_proxy_key  <<<N_TOK,                256, 0, stream>>>(x, idx, csum, proxy_n, key_n);
    k_gemm_qk    <<<dim3(16, 128),        128, 0, stream>>>(proxy_n, key_n, attn);
    k_softmax    <<<N_TOK,                256, 0, stream>>>(attn, P);
    k_gemm_pv    <<<dim3(48, 8),          128, 0, stream>>>(P, vT, out);
}
